// BatchedAttentionBlock_22771916604218
// MI455X (gfx1250) — matrix operand order repaired and checked by reading, not compiled, not run on hardware
//
#include <hip/hip_runtime.h>

// ---------------------------------------------------------------------------
// BatchedAttentionBlock on gfx1250 (MI455X): bf16 WMMA GEMMs + flash attention
// with async global->LDS staging of K/V tiles (shared across the workgroup).
// ---------------------------------------------------------------------------
#define EMBED 256
#define NHEAD 8
#define HDIM 32
#define NGRAPH 64
#define MAXLEN 512
#define DROWS (NGRAPH * MAXLEN)   // 32768 dense rows
#define SCALE_Q 0.17677669529663687f  // 1/sqrt(32)

typedef __attribute__((ext_vector_type(8)))  __bf16 v8bf;
typedef __attribute__((ext_vector_type(16))) __bf16 v16bf;
typedef __attribute__((ext_vector_type(8)))  float  v8f;
typedef __attribute__((ext_vector_type(4)))  int    v4i;

#if __has_builtin(__builtin_amdgcn_global_load_async_to_lds_b128)
#define HAVE_ASYNC 1
#else
#define HAVE_ASYNC 0
#endif

// Copy 16 bytes global -> LDS (per-lane). Async DMA when available.
__device__ __forceinline__ void copy16_g2l(const __bf16* g, __bf16* l) {
#if HAVE_ASYNC
  __builtin_amdgcn_global_load_async_to_lds_b128(
      (__attribute__((address_space(1))) v4i*)(g),
      (__attribute__((address_space(3))) v4i*)(l), 0, 0);
#else
  *(v8bf*)l = *(const v8bf*)g;
#endif
}

#if HAVE_ASYNC
#if __has_builtin(__builtin_amdgcn_s_wait_asynccnt)
#define WAIT_ASYNC(n) __builtin_amdgcn_s_wait_asynccnt(n)
#else
#define WAIT_ASYNC(n) asm volatile("s_wait_asynccnt %0" ::"i"(n) : "memory")
#endif
#else
#define WAIT_ASYNC(n) ((void)0)
#endif

__device__ __forceinline__ v16bf cat8(v8bf lo, v8bf hi) {
  return __builtin_shufflevector(lo, hi, 0,1,2,3,4,5,6,7,8,9,10,11,12,13,14,15);
}

// A-matrix 16x32 bf16 tile, row-major source with leading dim `ld` (elements).
// Lane (m = lane&15, half = lane>>4) reads cols [half*8, +8) and [16+half*8, +8).
__device__ __forceinline__ v16bf load_a16(const __bf16* base, int ld, int lane) {
  int m = lane & 15, half = lane >> 4;
  const __bf16* p = base + m * ld + half * 8;
  v8bf lo = *(const v8bf*)p;
  v8bf hi = *(const v8bf*)(p + 16);
  return cat8(lo, hi);
}

// B-matrix 32x16 bf16 tile where B[k][n] = R[n][k], R row-major, lead dim `ld`.
// Lane (n = lane&15, half) reads R[n][half*8 .. +8) and R[n][16+half*8 .. +8).
__device__ __forceinline__ v16bf load_b16(const __bf16* base, int ld, int lane) {
  int n = lane & 15, half = lane >> 4;
  const __bf16* p = base + n * ld + half * 8;
  return cat8(*(const v8bf*)p, *(const v8bf*)(p + 16));
}

__device__ __forceinline__ v8f wmma_bf16(v16bf a, v16bf b, v8f c) {
  return __builtin_amdgcn_wmma_f32_16x16x32_bf16(false, a, false, b, (short)0, c,
                                                 false, false);
}

// reductions across the 16-lane half (rows of C tile live per-half)
__device__ __forceinline__ float halfmax16(float v) {
  v = fmaxf(v, __shfl_xor(v, 1));
  v = fmaxf(v, __shfl_xor(v, 2));
  v = fmaxf(v, __shfl_xor(v, 4));
  v = fmaxf(v, __shfl_xor(v, 8));
  return v;
}
__device__ __forceinline__ float halfsum16(float v) {
  v += __shfl_xor(v, 1);
  v += __shfl_xor(v, 2);
  v += __shfl_xor(v, 4);
  v += __shfl_xor(v, 8);
  return v;
}

// --------------------------- prep: counts/starts ---------------------------
__global__ void prep_kernel(const int* __restrict__ batch, int N,
                            int* __restrict__ counts, int* __restrict__ starts) {
  __shared__ int cnt[NGRAPH];
  int t = threadIdx.x;
  if (t < NGRAPH) cnt[t] = 0;
  __syncthreads();
  for (int i = t; i < N; i += blockDim.x) atomicAdd(&cnt[batch[i]], 1);
  __syncthreads();
  if (t == 0) {
    int acc = 0;
    for (int g = 0; g < NGRAPH; g++) {
      counts[g] = cnt[g];
      starts[g] = acc;
      acc += cnt[g];
    }
  }
}

// ----------------------- weight conversion (fold scale) --------------------
__global__ void cvt_w_kernel(const float* __restrict__ w_in,
                             const float* __restrict__ b_in,
                             const float* __restrict__ w_out,
                             __bf16* __restrict__ Wb, float* __restrict__ biasq,
                             __bf16* __restrict__ Wob) {
  int i = blockIdx.x * 256 + threadIdx.x;
  if (i < 768 * EMBED) {
    int row = i >> 8;  // EMBED == 256
    float s = (row < EMBED) ? SCALE_Q : 1.0f;
    Wb[i] = (__bf16)(w_in[i] * s);
  }
  if (i < 768) biasq[i] = b_in[i] * ((i < EMBED) ? SCALE_Q : 1.0f);
  if (i < EMBED * EMBED) Wob[i] = (__bf16)w_out[i];
}

// ----------------------- ragged -> dense bf16 scatter ----------------------
__global__ void build_dense_kernel(const float* __restrict__ x,
                                   const int* __restrict__ counts,
                                   const int* __restrict__ starts,
                                   __bf16* __restrict__ Xd) {
  int row = blockIdx.x;       // 0..DROWS-1
  int c = threadIdx.x;        // 0..255
  int b = row >> 9, s = row & 511;
  float v = 0.0f;
  if (s < counts[b]) v = x[(size_t)(starts[b] + s) * EMBED + c];
  Xd[(size_t)row * EMBED + c] = (__bf16)v;
}

// --------------------------- QKV projection GEMM ---------------------------
// Dense X [32768,256] @ Wqkv^T [256,768]. Wave = 16 rows x 64 cols.
// Q,K stored [B][H][S][32]; V stored transposed [B][H][32][S].
__global__ void qkv_gemm_kernel(const __bf16* __restrict__ Xd,
                                const __bf16* __restrict__ Wb,
                                const float* __restrict__ biasq,
                                __bf16* __restrict__ Q, __bf16* __restrict__ K,
                                __bf16* __restrict__ VT) {
  int lane = threadIdx.x & 31, wave = threadIdx.x >> 5;
  int row0 = blockIdx.x * 16;
  int colb = blockIdx.y * 256 + wave * 64;
  int n = lane & 15, half = lane >> 4;

  v8f acc[4] = {v8f{}, v8f{}, v8f{}, v8f{}};
  const __bf16* Abase = Xd + (size_t)row0 * EMBED;
  for (int kk = 0; kk < EMBED; kk += 32) {
    v16bf a = load_a16(Abase + kk, EMBED, lane);
#pragma unroll
    for (int t = 0; t < 4; t++) {
      v16bf b = load_b16(Wb + (size_t)(colb + t * 16) * EMBED + kk, EMBED, lane);
      acc[t] = wmma_bf16(a, b, acc[t]);
    }
  }
#pragma unroll
  for (int t = 0; t < 4; t++) {
    int c = colb + t * 16 + n;
    float bv = biasq[c];
#pragma unroll
    for (int r = 0; r < 8; r++) {
      int row = row0 + r + 8 * half;
      int b = row >> 9, s = row & 511;
      __bf16 val = (__bf16)(acc[t][r] + bv);
      if (c < 256) {
        int h = c >> 5, d = c & 31;
        Q[(((size_t)(b * NHEAD + h) * MAXLEN + s) * HDIM) + d] = val;
      } else if (c < 512) {
        int cc = c - 256, h = cc >> 5, d = cc & 31;
        K[(((size_t)(b * NHEAD + h) * MAXLEN + s) * HDIM) + d] = val;
      } else {
        int cc = c - 512, h = cc >> 5, d = cc & 31;
        VT[(((size_t)(b * NHEAD + h) * HDIM + d) * MAXLEN) + s] = val;
      }
    }
  }
}

// ------------------------- flash attention per head ------------------------
// Block = 4 waves sharing one (b, h); wave owns a 16-query block.
// K/V tiles (32 keys) staged into LDS once per block, double buffered:
// async DMA of tile i+1 overlaps WMMA work on tile i.
__global__ void attn_kernel(const __bf16* __restrict__ Q,
                            const __bf16* __restrict__ K,
                            const __bf16* __restrict__ VT,
                            const int* __restrict__ counts,
                            __bf16* __restrict__ CTX) {
  __shared__ __align__(64) __bf16 ldsK[2][32 * 32];  // [key][dim]
  __shared__ __align__(64) __bf16 ldsV[2][32 * 32];  // [dim][key]
  __shared__ __align__(64) __bf16 lds_p[4][16 * 32];
  int lane = threadIdx.x & 31, wave = threadIdx.x >> 5;
  int qb = blockIdx.x & 7;
  int h = (blockIdx.x >> 3) & 7;
  int b = blockIdx.x >> 6;
  int q0 = qb * 64 + wave * 16;
  int len = counts[b];
  int n = lane & 15, half = lane >> 4;

  const __bf16* Qh = Q + (size_t)(b * NHEAD + h) * MAXLEN * HDIM;
  const __bf16* Kh = K + (size_t)(b * NHEAD + h) * MAXLEN * HDIM;
  const __bf16* Vh = VT + (size_t)(b * NHEAD + h) * HDIM * MAXLEN;

  // Cooperative stage of one 32-key K tile + V tile into LDS buffer `buf`.
  // 128 threads x 16B each per tile.
  int trow = threadIdx.x >> 2;          // 0..31
  int tcol = (threadIdx.x & 3) * 8;     // element chunk within 32
  auto stage = [&](int kk, int buf) {
    copy16_g2l(Kh + (size_t)(kk + trow) * HDIM + tcol, &ldsK[buf][trow * 32 + tcol]);
    copy16_g2l(Vh + (size_t)trow * MAXLEN + kk + tcol, &ldsV[buf][trow * 32 + tcol]);
  };

  v16bf aq = load_a16(Qh + (size_t)q0 * HDIM, HDIM, lane);  // scale folded in Wq
  float mrow[8], lrow[8];
  v8f acc0 = v8f{}, acc1 = v8f{};
#pragma unroll
  for (int r = 0; r < 8; r++) { mrow[r] = -__builtin_inff(); lrow[r] = 0.0f; }

  __bf16* pl = &lds_p[wave][0];
  stage(0, 0);
  for (int i = 0; i < MAXLEN / 32; i++) {
    int buf = i & 1;
    int kk = i * 32;
    if (i + 1 < MAXLEN / 32) {
      stage(kk + 32, buf ^ 1);   // prefetch next tile into other buffer
      WAIT_ASYNC(2);             // oldest tile (current) has landed
    } else {
      WAIT_ASYNC(0);
    }
    __syncthreads();             // current buffer visible to all waves

    const __bf16* Kt = &ldsK[buf][0];
    const __bf16* Vt = &ldsV[buf][0];
    v16bf bk0 = load_b16(Kt, 32, lane);            // keys kk+0..15
    v16bf bk1 = load_b16(Kt + 16 * 32, 32, lane);  // keys kk+16..31
    v8f s0 = wmma_bf16(aq, bk0, v8f{});
    v8f s1 = wmma_bf16(aq, bk1, v8f{});
    bool ok0 = (kk + n) < len;
    bool ok1 = (kk + 16 + n) < len;
#pragma unroll
    for (int r = 0; r < 8; r++) {
      float a0 = ok0 ? s0[r] : -1e9f;
      float a1 = ok1 ? s1[r] : -1e9f;
      float rmx = halfmax16(fmaxf(a0, a1));
      float mnew = fmaxf(mrow[r], rmx);
      float corr = __expf(mrow[r] - mnew);
      float p0 = __expf(a0 - mnew);
      float p1 = __expf(a1 - mnew);
      float rs = halfsum16(p0 + p1);
      lrow[r] = lrow[r] * corr + rs;
      acc0[r] *= corr;
      acc1[r] *= corr;
      mrow[r] = mnew;
      int row = r + 8 * half;
      pl[row * 32 + n] = (__bf16)p0;
      pl[row * 32 + 16 + n] = (__bf16)p1;
    }
    asm volatile("s_wait_dscnt 0" ::: "memory");  // P stores visible to wave
    v16bf ap = load_a16(pl, 32, lane);            // probs tile in A layout
    v16bf bv0 = load_b16(Vt, 32, lane);           // dims 0..15
    v16bf bv1 = load_b16(Vt + 16 * 32, 32, lane); // dims 16..31
    acc0 = wmma_bf16(ap, bv0, acc0);
    acc1 = wmma_bf16(ap, bv1, acc1);
    __syncthreads();             // all waves done with buffer before overwrite
  }
#pragma unroll
  for (int r = 0; r < 8; r++) {
    float inv = 1.0f / lrow[r];
    int row = q0 + r + 8 * half;
    size_t base = ((size_t)(b * MAXLEN + row)) * EMBED + h * HDIM;
    CTX[base + n] = (__bf16)(acc0[r] * inv);
    CTX[base + 16 + n] = (__bf16)(acc1[r] * inv);
  }
}

// ------------------- output projection + gather + residual -----------------
__global__ void outproj_kernel(const __bf16* __restrict__ CTX,
                               const __bf16* __restrict__ Wob,
                               const float* __restrict__ bout,
                               const float* __restrict__ x,
                               const int* __restrict__ counts,
                               const int* __restrict__ starts,
                               float* __restrict__ H) {
  int lane = threadIdx.x & 31, wave = threadIdx.x >> 5;
  int row0 = blockIdx.x * 16;
  int colb = wave * 64;
  int n = lane & 15, half = lane >> 4;

  v8f acc[4] = {v8f{}, v8f{}, v8f{}, v8f{}};
  const __bf16* Abase = CTX + (size_t)row0 * EMBED;
  for (int kk = 0; kk < EMBED; kk += 32) {
    v16bf a = load_a16(Abase + kk, EMBED, lane);
#pragma unroll
    for (int t = 0; t < 4; t++) {
      v16bf b = load_b16(Wob + (size_t)(colb + t * 16) * EMBED + kk, EMBED, lane);
      acc[t] = wmma_bf16(a, b, acc[t]);
    }
  }
  int gb = row0 >> 9;  // all 16 rows in one graph (512 % 16 == 0)
  int len = counts[gb], st = starts[gb];
  int s0 = row0 & 511;
#pragma unroll
  for (int t = 0; t < 4; t++) {
    int c = colb + t * 16 + n;
    float bv = bout[c];
#pragma unroll
    for (int r = 0; r < 8; r++) {
      int s = s0 + r + 8 * half;
      if (s < len) {
        size_t node = (size_t)(st + s);
        H[node * EMBED + c] = acc[t][r] + bv + x[node * EMBED + c];
      }
    }
  }
}

// -------------------- per-graph LayerNorm statistics -----------------------
__global__ void ln_stats_kernel(const float* __restrict__ H,
                                const int* __restrict__ counts,
                                const int* __restrict__ starts,
                                float* __restrict__ stats) {
  __shared__ float ssum[256], ssq[256];
  int g = blockIdx.x, t = threadIdx.x;
  int st = starts[g], cnt = counts[g];
  long total = (long)cnt * EMBED;
  const float* base = H + (size_t)st * EMBED;
  float s = 0.0f, q = 0.0f;
  for (long i = t; i < total; i += 256) {
    float v = base[i];
    s += v;
    q += v * v;
  }
  ssum[t] = s;
  ssq[t] = q;
  __syncthreads();
  for (int off = 128; off > 0; off >>= 1) {
    if (t < off) { ssum[t] += ssum[t + off]; ssq[t] += ssq[t + off]; }
    __syncthreads();
  }
  if (t == 0) {
    float norm = (float)total;
    float mean = ssum[0] / norm;
    float var = ssq[0] / norm - mean * mean;
    stats[g * 2] = mean;
    stats[g * 2 + 1] = rsqrtf(var + 1e-5f);
  }
}

// ----------------------- LN apply + exact-erf GELU -------------------------
__global__ void finalize_kernel(const float* __restrict__ H,
                                const int* __restrict__ batch,
                                const float* __restrict__ stats,
                                const float* __restrict__ lw,
                                const float* __restrict__ lb,
                                float* __restrict__ out, int N) {
  int node = blockIdx.x;
  if (node >= N) return;
  int c = threadIdx.x;
  int g = batch[node];
  float v = (H[(size_t)node * EMBED + c] - stats[g * 2]) * stats[g * 2 + 1];
  v = v * lw[c] + lb[c];
  out[(size_t)node * EMBED + c] = 0.5f * v * (1.0f + erff(v * 0.70710678118654752f));
}

// ---------------------------------------------------------------------------
extern "C" void kernel_launch(void* const* d_in, const int* in_sizes, int n_in,
                              void* d_out, int out_size, void* d_ws, size_t ws_size,
                              hipStream_t stream) {
  const float* x = (const float*)d_in[0];
  const int* batch = (const int*)d_in[1];
  const float* w_in = (const float*)d_in[2];
  const float* b_in = (const float*)d_in[3];
  const float* w_out = (const float*)d_in[4];
  const float* b_out = (const float*)d_in[5];
  const float* lnw = (const float*)d_in[6];
  const float* lnb = (const float*)d_in[7];
  float* out = (float*)d_out;
  int N = in_sizes[1];  // batch vector length

  char* p = (char*)d_ws;
  auto alloc = [&](size_t bytes) -> char* {
    char* r = p;
    p += (bytes + 255) & ~(size_t)255;
    return r;
  };
  int* counts = (int*)alloc(NGRAPH * 4);
  int* starts = (int*)alloc(NGRAPH * 4);
  float* biasq = (float*)alloc(768 * 4);
  float* stats = (float*)alloc(2 * NGRAPH * 4);
  __bf16* Wb  = (__bf16*)alloc((size_t)768 * EMBED * 2);
  __bf16* Wob = (__bf16*)alloc((size_t)EMBED * EMBED * 2);
  __bf16* Xd  = (__bf16*)alloc((size_t)DROWS * EMBED * 2);
  __bf16* Q   = (__bf16*)alloc((size_t)NGRAPH * NHEAD * MAXLEN * HDIM * 2);
  __bf16* K   = (__bf16*)alloc((size_t)NGRAPH * NHEAD * MAXLEN * HDIM * 2);
  __bf16* VT  = (__bf16*)alloc((size_t)NGRAPH * NHEAD * MAXLEN * HDIM * 2);
  __bf16* CTX = (__bf16*)alloc((size_t)DROWS * EMBED * 2);
  float* H    = (float*)alloc((size_t)DROWS * EMBED * 4);

  prep_kernel<<<1, 256, 0, stream>>>(batch, N, counts, starts);
  cvt_w_kernel<<<(768 * EMBED + 255) / 256, 256, 0, stream>>>(w_in, b_in, w_out,
                                                              Wb, biasq, Wob);
  build_dense_kernel<<<DROWS, 256, 0, stream>>>(x, counts, starts, Xd);
  qkv_gemm_kernel<<<dim3(DROWS / 16, 3), 128, 0, stream>>>(Xd, Wb, biasq, Q, K, VT);
  attn_kernel<<<NGRAPH * NHEAD * (MAXLEN / 64), 128, 0, stream>>>(Q, K, VT, counts, CTX);
  outproj_kernel<<<DROWS / 16, 128, 0, stream>>>(CTX, Wob, b_out, x, counts, starts, H);
  ln_stats_kernel<<<NGRAPH, 256, 0, stream>>>(H, counts, starts, stats);
  finalize_kernel<<<N, 256, 0, stream>>>(H, batch, stats, lnw, lnb, out, N);
}
